// RNN_14173392077294
// MI455X (gfx1250) — compile-verified, matrix operand order checked by reading, not run on hardware
//
#include <hip/hip_runtime.h>
#include <cmath>

// Problem sizes (match reference)
#define Tn 4096
#define Dn 768
#define Hn 1462
#define Hp 1472   // H padded to multiple of 16 for WMMA tiles / b128 strides
#define Cn 19

#define SCAN_BLOCKS  64
#define SCAN_THREADS 512

typedef __attribute__((ext_vector_type(16))) __bf16 v16bf;
typedef __attribute__((ext_vector_type(8)))  float  v8f;

// ---------------------------------------------------------------------------
// Prologue packs
// ---------------------------------------------------------------------------
// elementwise f32 -> bf16 (row-major copy)
__global__ void __launch_bounds__(256)
f32_to_bf16_k(const float* __restrict__ src, __bf16* __restrict__ dst, int n) {
  int i = blockIdx.x * 256 + threadIdx.x;
  if (i < n) dst[i] = (__bf16)src[i];
}

// pad rows: dst (rows x pcols, f32) = src (rows x ncols) with zero pad cols
__global__ void __launch_bounds__(256)
pad_rows_k(const float* __restrict__ src, float* __restrict__ dst,
           int rows, int ncols, int pcols) {
  int i = blockIdx.x * 256 + threadIdx.x;
  if (i >= rows * pcols) return;
  int r = i / pcols, c = i - r * pcols;
  dst[i] = (c < ncols) ? src[(size_t)r * ncols + c] : 0.f;
}

// ---------------------------------------------------------------------------
// WMMA GEMM: P1[t][h] = sum_k X[t][k] * W1[h][k]  + (b_ih1[h] + b_hh1[h])
// bf16 operands pre-packed; 2 b128 loads per operand per WMMA; f32 accum.
// One wave per 16x16 output tile; K loop over D=768 in steps of 32, unroll 4.
// ---------------------------------------------------------------------------
union Bf16Vec { uint4 q[2]; v16bf v; };

__global__ void __launch_bounds__(32)
gemm_p1_wmma(const __bf16* __restrict__ X, const __bf16* __restrict__ W,
             const float* __restrict__ bi, const float* __restrict__ bh,
             float* __restrict__ P) {
  const int lane = threadIdx.x;           // 0..31
  const int m    = lane & 15;
  const int koff = (lane >> 4) * 8;       // 16-bit A/B layout: half-wave K offset
  const int tileM = blockIdx.x * 16;      // over T
  const int tileN = blockIdx.y * 16;      // over Hp

  const int rowA = tileM + m;             // X row (always < Tn)
  int rowB = tileN + m;                   // W row (n index), clamp for pad tiles
  if (rowB >= Hn) rowB = Hn - 1;

  const __bf16* __restrict__ xrow = X + (size_t)rowA * Dn;
  const __bf16* __restrict__ wrow = W + (size_t)rowB * Dn;

  v8f acc = {0.f, 0.f, 0.f, 0.f, 0.f, 0.f, 0.f, 0.f};

#pragma unroll 4
  for (int k0 = 0; k0 < Dn; k0 += 32) {
    // element i of v16bf holds K = (i<8 ? 0 : 16) + koff + (i&7)
    Bf16Vec a, b;
    a.q[0] = *(const uint4*)(xrow + k0 + koff);        // 8 bf16 = 16B, aligned
    a.q[1] = *(const uint4*)(xrow + k0 + 16 + koff);
    b.q[0] = *(const uint4*)(wrow + k0 + koff);
    b.q[1] = *(const uint4*)(wrow + k0 + 16 + koff);

    acc = __builtin_amdgcn_wmma_f32_16x16x32_bf16(
        /*neg_a=*/false, a.v, /*neg_b=*/false, b.v,
        /*c_mod=*/(short)0, acc, /*reuse_a=*/false, /*reuse_b=*/false);
  }

  // C/D layout: VGPR r -> M = r + 8*(lane>>4), N = lane&15
  const int mc = (lane >> 4) * 8;
  const int nc = lane & 15;
  const int nG = tileN + nc;
  if (nG < Hn) {
    const float bias = bi[nG] + bh[nG];
#pragma unroll
    for (int r = 0; r < 8; ++r)
      P[(size_t)(tileM + mc + r) * Hp + nG] = acc[r] + bias;
  } else {
#pragma unroll
    for (int r = 0; r < 8; ++r)
      P[(size_t)(tileM + mc + r) * Hp + nG] = 0.f;  // defined pad
  }
}

// ---------------------------------------------------------------------------
// Wave-parallel dot products (wave32), result valid in lane 0.
// ---------------------------------------------------------------------------
// float2 version for odd/unpadded strides (b64 loads, always aligned here)
__device__ __forceinline__ float wave_dot(const float* __restrict__ a,
                                          const float* __restrict__ b,
                                          int n, int lane) {
  float s = 0.f;
  const int n2 = n >> 1;
  const float2* a2 = (const float2*)a;
  const float2* b2 = (const float2*)b;
  for (int i = lane; i < n2; i += 32) {
    float2 va = a2[i], vb = b2[i];
    s = fmaf(va.x, vb.x, s);
    s = fmaf(va.y, vb.y, s);
  }
  if ((n & 1) && lane == 0) s += a[n - 1] * b[n - 1];
#pragma unroll
  for (int off = 16; off > 0; off >>= 1) s += __shfl_down(s, off, 32);
  return s;
}

// float4 version: requires n % 4 == 0 and 16B-aligned pointers (b128 loads)
__device__ __forceinline__ float wave_dot4(const float* __restrict__ a,
                                           const float* __restrict__ b,
                                           int n, int lane) {
  float s = 0.f;
  const int n4 = n >> 2;
  const float4* a4 = (const float4*)a;
  const float4* b4 = (const float4*)b;
  for (int i = lane; i < n4; i += 32) {
    float4 va = a4[i], vb = b4[i];
    s = fmaf(va.x, vb.x, s);
    s = fmaf(va.y, vb.y, s);
    s = fmaf(va.z, vb.z, s);
    s = fmaf(va.w, vb.w, s);
  }
#pragma unroll
  for (int off = 16; off > 0; off >>= 1) s += __shfl_down(s, off, 32);
  return s;
}

// ---------------------------------------------------------------------------
// Software grid barrier (all SCAN_BLOCKS co-resident; L2 atomics + s_sleep).
// ---------------------------------------------------------------------------
__device__ __forceinline__ void grid_barrier(int* cnt, int* gen, int nb) {
  __syncthreads();
  if (threadIdx.x == 0) {
    __threadfence();
    int g = __hip_atomic_load(gen, __ATOMIC_RELAXED, __HIP_MEMORY_SCOPE_AGENT);
    int prev = __hip_atomic_fetch_add(cnt, 1, __ATOMIC_ACQ_REL, __HIP_MEMORY_SCOPE_AGENT);
    if (prev == nb - 1) {
      __hip_atomic_store(cnt, 0, __ATOMIC_RELAXED, __HIP_MEMORY_SCOPE_AGENT);
      __hip_atomic_fetch_add(gen, 1, __ATOMIC_RELEASE, __HIP_MEMORY_SCOPE_AGENT);
    } else {
      while (__hip_atomic_load(gen, __ATOMIC_ACQUIRE, __HIP_MEMORY_SCOPE_AGENT) == g)
        __builtin_amdgcn_s_sleep(2);
    }
    __threadfence();
  }
  __syncthreads();
}

__global__ void init_k(int* bar) { bar[0] = 0; bar[1] = 0; }

// ---------------------------------------------------------------------------
// Sequential recurrence. Per step:
//   h1_t = tanh(P1[t] + w_hh1 @ h1_{t-1})                (x-term precomputed)
//   h2_t = tanh(w_ih2 @ h1_t + w_hh2 @ h2_{t-1} + b)
// 1024 waves across 64 blocks, wave-per-row, b128 dots on padded weights;
// one grid barrier per step (phase-2/next-phase-1 ordering is transitive).
// ---------------------------------------------------------------------------
__global__ void __launch_bounds__(SCAN_THREADS)
scan_kernel(const float* __restrict__ P1, float* __restrict__ h1s,
            float* __restrict__ h2s,
            const float* __restrict__ w_hh1p,   // Hn x Hp, zero-padded
            const float* __restrict__ w_ih2p,   // Dn x Hp, zero-padded
            const float* __restrict__ w_hh2,    // Dn x Dn (native, b128-clean)
            const float* __restrict__ b_ih2, const float* __restrict__ b_hh2,
            int* bar_cnt, int* bar_gen) {
  const int lane = threadIdx.x & 31;
  const int wave = blockIdx.x * (SCAN_THREADS / 32) + (threadIdx.x >> 5);
  const int NW = SCAN_BLOCKS * (SCAN_THREADS / 32);  // 1024 waves

  for (int t = 0; t < Tn; ++t) {
    const float* h1prev = h1s + (size_t)(t - 1) * Hp;   // used only for t>0
    // ---- phase 1: h1 rows (also zero the pad columns of this step's row) ----
    for (int r = wave; r < Hp; r += NW) {
      if (r < Hn) {
        if (r + NW < Hn)  // warm L0 for the next row this wave will process
          __builtin_prefetch((const void*)(w_hh1p + (size_t)(r + NW) * Hp + lane * 32), 0, 3);
        float d = (t > 0) ? wave_dot4(w_hh1p + (size_t)r * Hp, h1prev, Hp, lane) : 0.f;
        if (lane == 0)
          h1s[(size_t)t * Hp + r] = tanhf(P1[(size_t)t * Hp + r] + d);
      } else if (lane == 0) {
        h1s[(size_t)t * Hp + r] = 0.f;   // pad col: keeps padded-length dots exact
      }
    }
    grid_barrier(bar_cnt, bar_gen, SCAN_BLOCKS);
    // ---- phase 2: h2 rows ----
    const float* h1cur  = h1s + (size_t)t * Hp;
    const float* h2prev = h2s + (size_t)(t - 1) * Dn;
    for (int r = wave; r < Dn; r += NW) {
      float d1 = wave_dot4(w_ih2p + (size_t)r * Hp, h1cur, Hp, lane);
      float d2 = (t > 0) ? wave_dot4(w_hh2 + (size_t)r * Dn, h2prev, Dn, lane) : 0.f;
      if (lane == 0)
        h2s[(size_t)t * Dn + r] = tanhf(d1 + d2 + b_ih2[r] + b_hh2[r]);
    }
  }
}

// ---------------------------------------------------------------------------
// Generic wave-per-row matvec:
//   y[i] = act( scale * ( A[i]·x (N) + A2[i]·x2 (N2) + bias[i] + bias2[i] + *sAdd ) )
// ---------------------------------------------------------------------------
__global__ void __launch_bounds__(256)
matvec_k(const float* __restrict__ A, int lda, const float* __restrict__ x,
         const float* __restrict__ bias,
         const float* __restrict__ A2, int lda2, const float* __restrict__ x2,
         const float* __restrict__ bias2,
         const float* __restrict__ sAdd, float scale, int act,
         float* __restrict__ y, int M, int N, int N2) {
  const int lane = threadIdx.x & 31;
  const int row = blockIdx.x * 8 + (threadIdx.x >> 5);
  if (row >= M) return;
  float s = wave_dot(A + (size_t)row * lda, x, N, lane);
  if (A2) s += wave_dot(A2 + (size_t)row * lda2, x2, N2, lane);
  if (lane == 0) {
    if (bias)  s += bias[row];
    if (bias2) s += bias2[row];
    if (sAdd)  s += *sAdd;
    s *= scale;
    if (act) s = tanhf(s);
    y[row] = s;
  }
}

// y[c] = sum_r x[r] * A[r*lda + c]   (column-weighted sum; coalesced over c)
__global__ void __launch_bounds__(256)
colsum_k(const float* __restrict__ A, int lda, const float* __restrict__ x,
         float* __restrict__ y, int rows, int cols) {
  const int c = blockIdx.x * blockDim.x + threadIdx.x;
  if (c >= cols) return;
  float s = 0.f;
  for (int r = 0; r < rows; ++r) s = fmaf(x[r], A[(size_t)r * lda + c], s);
  y[c] = s;
}

// In-place softmax over n elements, single block.
__global__ void __launch_bounds__(1024)
softmax_k(float* __restrict__ s, int n) {
  __shared__ float red[32];
  const int tid = threadIdx.x, lane = tid & 31, wid = tid >> 5;
  float m = -3.402823e38f;
  for (int i = tid; i < n; i += 1024) m = fmaxf(m, s[i]);
#pragma unroll
  for (int off = 16; off > 0; off >>= 1) m = fmaxf(m, __shfl_down(m, off, 32));
  if (lane == 0) red[wid] = m;
  __syncthreads();
  if (wid == 0) {
    float v = red[lane];
#pragma unroll
    for (int off = 16; off > 0; off >>= 1) v = fmaxf(v, __shfl_down(v, off, 32));
    if (lane == 0) red[0] = v;
  }
  __syncthreads();
  m = red[0];
  __syncthreads();
  float sum = 0.f;
  for (int i = tid; i < n; i += 1024) { float e = __expf(s[i] - m); s[i] = e; sum += e; }
#pragma unroll
  for (int off = 16; off > 0; off >>= 1) sum += __shfl_down(sum, off, 32);
  if (lane == 0) red[wid] = sum;
  __syncthreads();
  if (wid == 0) {
    float v = red[lane];
#pragma unroll
    for (int off = 16; off > 0; off >>= 1) v += __shfl_down(v, off, 32);
    if (lane == 0) red[0] = v;
  }
  __syncthreads();
  const float inv = 1.f / red[0];
  for (int i = tid; i < n; i += 1024) s[i] *= inv;
}

// ---------------------------------------------------------------------------
extern "C" void kernel_launch(void* const* d_in, const int* in_sizes, int n_in,
                              void* d_out, int out_size, void* d_ws, size_t ws_size,
                              hipStream_t stream) {
  (void)in_sizes; (void)n_in; (void)out_size; (void)ws_size;
  const float* xin    = (const float*)d_in[0];
  const float* w_ih1  = (const float*)d_in[1];
  const float* w_hh1  = (const float*)d_in[2];
  const float* b_ih1  = (const float*)d_in[3];
  const float* b_hh1  = (const float*)d_in[4];
  const float* w_ih2  = (const float*)d_in[5];
  const float* w_hh2  = (const float*)d_in[6];
  const float* b_ih2  = (const float*)d_in[7];
  const float* b_hh2  = (const float*)d_in[8];
  const float* in_w1  = (const float*)d_in[9];
  const float* in_b1  = (const float*)d_in[10];
  const float* out_w1 = (const float*)d_in[11];
  const float* out_b1 = (const float*)d_in[12];
  const float* in_w2  = (const float*)d_in[13];
  const float* in_b2  = (const float*)d_in[14];
  const float* out_w2 = (const float*)d_in[15];
  const float* out_b2 = (const float*)d_in[16];
  const float* cls_w  = (const float*)d_in[17];
  const float* cls_b  = (const float*)d_in[18];
  float* out = (float*)d_out;

  // ---- workspace carve-up (~83 MB) ----
  char* ws = (char*)d_ws;
  size_t off = 0;
  auto takeB = [&](size_t bytes) -> void* {
    void* p = (void*)(ws + off);
    off = (off + bytes + 255) & ~(size_t)255;
    return p;
  };
  auto take = [&](size_t elems) -> float* { return (float*)takeB(elems * 4); };

  float*  P1     = take((size_t)Tn * Hp);
  float*  h1s    = take((size_t)Tn * Hp);
  float*  h2s    = take((size_t)Tn * Dn);
  float*  w_hh1p = take((size_t)Hn * Hp);           // padded f32 weights
  float*  w_ih2p = take((size_t)Dn * Hp);
  __bf16* Xbf    = (__bf16*)takeB((size_t)Tn * Dn * 2);  // bf16 GEMM operands
  __bf16* W1bf   = (__bf16*)takeB((size_t)Hn * Dn * 2);
  float* q1   = take(Hp);  float* u1  = take(Hp);  float* c1  = take(Hp);
  float* av1  = take(Hp);  float* m1v = take(Hp);  float* sc1 = take(Tn);
  float* q2   = take(Dn);  float* u2  = take(Dn);  float* c2  = take(Dn);
  float* av2  = take(Dn);  float* m2v = take(Dn);  float* sc2 = take(Tn);
  float* qbk1 = take(64);  float* qbk2 = take(64);
  float* h1f  = take(Hp);  float* h2f = take(Dn);
  int*   bar  = (int*)take(64);

  const float inv_sqrtH = 1.0f / sqrtf((float)Hn);
  const float inv_sqrtD = 1.0f / sqrtf((float)Dn);

  // 0) barrier init + operand packing (independent prologue)
  init_k<<<1, 1, 0, stream>>>(bar);
  f32_to_bf16_k<<<(Tn * Dn + 255) / 256, 256, 0, stream>>>(xin, Xbf, Tn * Dn);
  f32_to_bf16_k<<<(Hn * Dn + 255) / 256, 256, 0, stream>>>(w_ih1, W1bf, Hn * Dn);
  pad_rows_k<<<(Hn * Hp + 255) / 256, 256, 0, stream>>>(w_hh1, w_hh1p, Hn, Hn, Hp);
  pad_rows_k<<<(Dn * Hp + 255) / 256, 256, 0, stream>>>(w_ih2, w_ih2p, Dn, Hn, Hp);

  // 1) P1 = x @ w_ih1^T + (b_ih1 + b_hh1)  — WMMA bf16, f32 accum
  gemm_p1_wmma<<<dim3(Tn / 16, Hp / 16), 32, 0, stream>>>(Xbf, W1bf, b_ih1, b_hh1, P1);

  // 2) sequential scan (persistent, grid-barrier per step)
  scan_kernel<<<SCAN_BLOCKS, SCAN_THREADS, 0, stream>>>(
      P1, h1s, h2s, w_hh1p, w_ih2p, w_hh2, b_ih2, b_hh2, bar, bar + 1);

  const float* h1last = h1s + (size_t)(Tn - 1) * Hp;
  const float* h2last = h2s + (size_t)(Tn - 1) * Dn;

  // ---- attention head 1 (E = H), last query only ----
  matvec_k<<<(Hn + 7) / 8, 256, 0, stream>>>(in_w1, Hn, h1last, in_b1,
      nullptr, 0, nullptr, nullptr, nullptr, 1.f, 0, q1, Hn, Hn, 0);
  matvec_k<<<1, 256, 0, stream>>>(q1, Hn, in_b1 + Hn, nullptr,
      nullptr, 0, nullptr, nullptr, nullptr, 1.f, 0, qbk1, 1, Hn, 0);      // q·bk
  colsum_k<<<(Hn + 255) / 256, 256, 0, stream>>>(in_w1 + (size_t)Hn * Hn, Hn, q1,
      u1, Hn, Hn);                                                         // u = Wk^T q
  matvec_k<<<(Tn + 7) / 8, 256, 0, stream>>>(h1s, Hp, u1, nullptr,
      nullptr, 0, nullptr, nullptr, qbk1, inv_sqrtH, 0, sc1, Tn, Hn, 0);   // scores
  softmax_k<<<1, 1024, 0, stream>>>(sc1, Tn);
  colsum_k<<<(Hn + 255) / 256, 256, 0, stream>>>(h1s, Hp, sc1, c1, Tn, Hn); // ctx
  matvec_k<<<(Hn + 7) / 8, 256, 0, stream>>>(in_w1 + (size_t)2 * Hn * Hn, Hn, c1,
      in_b1 + 2 * Hn, nullptr, 0, nullptr, nullptr, nullptr, 1.f, 0, av1, Hn, Hn, 0);
  matvec_k<<<(Hn + 7) / 8, 256, 0, stream>>>(out_w1, Hn, av1, out_b1,
      nullptr, 0, nullptr, nullptr, nullptr, 1.f, 0, m1v, Hn, Hn, 0);

  // ---- attention head 2 (E = D) ----
  matvec_k<<<(Dn + 7) / 8, 256, 0, stream>>>(in_w2, Dn, h2last, in_b2,
      nullptr, 0, nullptr, nullptr, nullptr, 1.f, 0, q2, Dn, Dn, 0);
  matvec_k<<<1, 256, 0, stream>>>(q2, Dn, in_b2 + Dn, nullptr,
      nullptr, 0, nullptr, nullptr, nullptr, 1.f, 0, qbk2, 1, Dn, 0);
  colsum_k<<<(Dn + 255) / 256, 256, 0, stream>>>(in_w2 + (size_t)Dn * Dn, Dn, q2,
      u2, Dn, Dn);
  matvec_k<<<(Tn + 7) / 8, 256, 0, stream>>>(h2s, Dn, u2, nullptr,
      nullptr, 0, nullptr, nullptr, qbk2, inv_sqrtD, 0, sc2, Tn, Dn, 0);
  softmax_k<<<1, 1024, 0, stream>>>(sc2, Tn);
  colsum_k<<<(Dn + 255) / 256, 256, 0, stream>>>(h2s, Dn, sc2, c2, Tn, Dn);
  matvec_k<<<(Dn + 7) / 8, 256, 0, stream>>>(in_w2 + (size_t)2 * Dn * Dn, Dn, c2,
      in_b2 + 2 * Dn, nullptr, 0, nullptr, nullptr, nullptr, 1.f, 0, av2, Dn, Dn, 0);
  matvec_k<<<(Dn + 7) / 8, 256, 0, stream>>>(out_w2, Dn, av2, out_b2,
      nullptr, 0, nullptr, nullptr, nullptr, 1.f, 0, m2v, Dn, Dn, 0);

  // ---- final cells + classifier ----
  const float* xlast = xin + (size_t)(Tn - 1) * Dn;
  matvec_k<<<(Hn + 7) / 8, 256, 0, stream>>>(w_ih1, Dn, xlast, b_ih1,
      w_hh1, Hn, m1v, b_hh1, nullptr, 1.f, 1, h1f, Hn, Dn, Hn);   // tanh
  matvec_k<<<(Dn + 7) / 8, 256, 0, stream>>>(w_ih2, Hn, h1f, b_ih2,
      w_hh2, Dn, m2v, b_hh2, nullptr, 1.f, 1, h2f, Dn, Hn, Dn);   // tanh
  matvec_k<<<(Cn + 7) / 8, 256, 0, stream>>>(cls_w, Dn, h2f, cls_b,
      nullptr, 0, nullptr, nullptr, nullptr, 1.f, 0, out, Cn, Dn, 0);
}
